// CLOSEgaps_76227079569583
// MI455X (gfx1250) — compile-verified, matrix operand order checked by reading, not run on hardware
//
#include <hip/hip_runtime.h>
#include <hip/hip_bf16.h>

typedef _Float16 half4  __attribute__((ext_vector_type(4)));
typedef _Float16 half8  __attribute__((ext_vector_type(8)));
typedef _Float16 half16 __attribute__((ext_vector_type(16)));
typedef float    f32x8  __attribute__((ext_vector_type(8)));
typedef int      v4i    __attribute__((ext_vector_type(4)));

#define WMMA_F32_F16(a, b, c) \
    __builtin_amdgcn_wmma_f32_16x16x32_f16(false, (a), false, (b), (short)0, (c), false, false)

#if defined(__gfx1250__) && __has_builtin(__builtin_amdgcn_sched_barrier)
#define SCHED_FENCE() __builtin_amdgcn_sched_barrier(0)
#else
#define SCHED_FENCE()
#endif

// ---------------------------------------------------------------------------
// gfx1250 async global->LDS path (ASYNCcnt-tracked DMA, no VGPR round trip).
// ---------------------------------------------------------------------------
#if defined(__gfx1250__) && __has_builtin(__builtin_amdgcn_global_load_async_to_lds_b128)
#define HAVE_ASYNC 1
#else
#define HAVE_ASYNC 0
#endif

#if HAVE_ASYNC
typedef __attribute__((address_space(1))) v4i* gv4i_p;
typedef __attribute__((address_space(3))) v4i* lv4i_p;
__device__ __forceinline__ void async_cp16(const void* g, void* l) {
    __builtin_amdgcn_global_load_async_to_lds_b128((gv4i_p)(g), (lv4i_p)(l), 0, 0);
}
__device__ __forceinline__ void wait_async0() {
#if __has_builtin(__builtin_amdgcn_s_wait_asynccnt)
    __builtin_amdgcn_s_wait_asynccnt(0);
#else
    asm volatile("s_wait_asynccnt 0x0" ::: "memory");
#endif
}
#endif

// ---------------------------------------------------------------------------
// Fragment loaders.  16-bit A-matrix 16x32 layout (ISA 7.12.2):
//   lanes 0-15 : row = lane,    halves 0..7 -> K=k0..k0+7,  halves 8..15 -> K=k0+16..k0+23
//   lanes 16-31: row = lane-16, halves 0..7 -> K=k0+8..+15, halves 8..15 -> K=k0+24..+31
// B operand uses the same pattern on the transposed matrix (row = output col).
// ---------------------------------------------------------------------------
__device__ __forceinline__ half16 gfrag(const _Float16* base, int ld, int row,
                                        int k0, int lane) {
    const _Float16* p = base + (size_t)row * ld + k0 + ((lane & 16) ? 8 : 0);
    half8 lo = *(const half8*)p;
    half8 hi = *(const half8*)(p + 16);
    return __builtin_shufflevector(lo, hi, 0,1,2,3,4,5,6,7,8,9,10,11,12,13,14,15);
}

// LDS fragment load, padded row stride of 40 halves (80B) -> conflict-friendly.
__device__ __forceinline__ half16 lfrag(const _Float16* s, int row, int roff) {
    const _Float16* p = s + row * 40 + roff;
    half8 lo = *(const half8*)p;
    half8 hi = *(const half8*)(p + 16);
    return __builtin_shufflevector(lo, hi, 0,1,2,3,4,5,6,7,8,9,10,11,12,13,14,15);
}

// ---------------------------------------------------------------------------
// One-time prep kernels
// ---------------------------------------------------------------------------
__global__ void k_cvt_f16(const float* __restrict__ src, _Float16* __restrict__ dst, int n) {
    int i = blockIdx.x * 256 + threadIdx.x;
    if (i < n) dst[i] = (_Float16)src[i];
}

// src: [H][K][N] f32  ->  dst: [H][N][K] f16   (grid covers H*K*N exactly)
__global__ void k_transpose_cvt(const float* __restrict__ src, _Float16* __restrict__ dst,
                                int K, int N) {
    int idx = blockIdx.x * 256 + threadIdx.x;
    int h = idx / (K * N);
    int rem = idx - h * (K * N);
    int n = rem / K;
    int k = rem - n * K;
    dst[(size_t)h * N * K + (size_t)n * K + k] = (_Float16)src[(size_t)h * K * N + (size_t)k * N + n];
}

// incidence f32 [8192][4096] -> f16 copy + invdeg[row] = 1/(rowsum + 1e-8)
__global__ void __launch_bounds__(256) k_inc_prep(const float* __restrict__ inc,
                                                  _Float16* __restrict__ incf,
                                                  float* __restrict__ invdeg) {
    __shared__ float red[256];
    const int row = blockIdx.x, t = threadIdx.x;
    const float4* src = (const float4*)(inc + (size_t)row * 4096);
    half4* dst = (half4*)(incf + (size_t)row * 4096);
    float s = 0.f;
    #pragma unroll
    for (int it = 0; it < 4; ++it) {
        int j = t + it * 256;                 // float4 index, 1024 per row
        float4 v = src[j];
        half4 h; h.x = (_Float16)v.x; h.y = (_Float16)v.y;
        h.z = (_Float16)v.z; h.w = (_Float16)v.w;
        dst[j] = h;
        s += v.x + v.y + v.z + v.w;
    }
    red[t] = s; __syncthreads();
    for (int st = 128; st; st >>= 1) {
        if (t < st) red[t] += red[t + st];
        __syncthreads();
    }
    if (t == 0) invdeg[row] = 1.f / (red[0] + 1e-8f);
}

// ---------------------------------------------------------------------------
// Generic WMMA GEMM:  C[M,N] = A[M,K] (f16, row-major) * BT[N,K]^T + bias[N]
// Block = 256 threads (8 waves); MTILES M-tiles of 16 rows (BM = MTILES*16);
// 8/MTILES wave-groups across N, NT 16-col tiles per wave.
// Requires (8/MTILES) * NT * 16 == N.
// MODE 0: store f32 row-major into C.   MODE 1: store f16 transposed into CT (ld = ldT).
// ---------------------------------------------------------------------------
template <int MTILES, int NT, int MODE>
__global__ void __launch_bounds__(256)
k_gemm(const _Float16* __restrict__ A, const _Float16* __restrict__ BT,
       const float* __restrict__ bias, float* __restrict__ C,
       _Float16* __restrict__ CT, int M, int K, int N, int ldT) {
    constexpr int NGRP = 8 / MTILES;
    const int tid = threadIdx.x, lane = tid & 31, w = tid >> 5;
    const int mi = w / NGRP, nq = w % NGRP;
    const int m0 = blockIdx.x * (MTILES * 16) + mi * 16;
    const int n0 = nq * (NT * 16);
    const int roff = (lane & 16) ? 8 : 0;
    const int rl = lane & 15;

    f32x8 acc[NT];
    #pragma unroll
    for (int t = 0; t < NT; ++t)
        acc[t] = (f32x8){0.f,0.f,0.f,0.f,0.f,0.f,0.f,0.f};

    for (int k0 = 0; k0 < K; k0 += 32) {
        half16 a = gfrag(A, K, m0 + rl, k0, lane);
        half16 bf[NT];
        #pragma unroll
        for (int t = 0; t < NT; ++t)
            bf[t] = gfrag(BT, K, n0 + t * 16 + rl, k0, lane);
        SCHED_FENCE();           // issue all fragment loads before the WMMA chain
        #pragma unroll
        for (int t = 0; t < NT; ++t)
            acc[t] = WMMA_F32_F16(a, bf[t], acc[t]);
    }

    #pragma unroll
    for (int t = 0; t < NT; ++t) {
        const int col = n0 + t * 16 + rl;
        const float bv = bias[col];
        #pragma unroll
        for (int r = 0; r < 8; ++r) {
            const int row = m0 + roff + r;
            const float v = acc[t][r] + bv;
            if (MODE == 0) C[(size_t)row * N + col] = v;
            else           CT[(size_t)col * ldT + row] = (_Float16)v;
        }
    }
}

// dot[row] = tn[row,:] . wa   (one wave per row)
__global__ void __launch_bounds__(256) k_dot(const float* __restrict__ tn,
                                             const float* __restrict__ wa,
                                             float* __restrict__ dotv) {
    const int lane = threadIdx.x & 31, w = threadIdx.x >> 5;
    const int row = blockIdx.x * 8 + w;
    const float* p = tn + (size_t)row * 256;
    float s = 0.f;
    for (int j = lane; j < 256; j += 32) s += p[j] * wa[j];
    for (int off = 16; off; off >>= 1) s += __shfl_xor(s, off, 32);
    if (lane == 0) dotv[row] = s;
}

// ---------------------------------------------------------------------------
// Fused big kernel:  agg = (incidence_f16 @ te) * invdeg  (M=8192, K=4096, N=256)
// + score = leaky(dotTn + agg.wa + ba), coeff = sigmoid(score)
// + upd = coeff*agg + tn   -> f16
// BM=32 (256 blocks -> ~2 blocks/WGP on a ~128-WGP part), double-buffered LDS,
// async global->LDS staging, 1 barrier per K-step; waves: mi = w>>2, nq = w&3.
// ---------------------------------------------------------------------------
__global__ void __launch_bounds__(256)
k_agg(const _Float16* __restrict__ incf, const _Float16* __restrict__ teT,
      const float* __restrict__ invdeg, const float* __restrict__ tn,
      const float* __restrict__ dotTn, const float* __restrict__ wa,
      const float* __restrict__ ba_all, int head, _Float16* __restrict__ updf) {
    __shared__ __align__(16) _Float16 Ash[2][32 * 40];
    __shared__ __align__(16) _Float16 Bsh[2][256 * 40];
    __shared__ float part[32 * 65];          // stride 65 -> conflict-free reduce
    __shared__ float sdeg[32];
    __shared__ float swa[256];
    __shared__ float scoef[32];

    const int tid = threadIdx.x, lane = tid & 31, w = tid >> 5;
    const int mi = w >> 2, nq = w & 3;
    const int m0 = blockIdx.x * 32;
    const int n0 = nq * 64;
    const int roff = (lane & 16) ? 8 : 0;
    const int rl = lane & 15;

    swa[tid] = wa[tid];
    if (tid < 32) sdeg[tid] = invdeg[m0 + tid];

    const int arow = (tid >> 2) & 31, ac = tid & 3;   // A staged by threads 0..127

    // stage one 32x32 A-tile + 256x32 B-tile into buffer `buf`
    auto stage = [&](int buf, int k0) {
#if HAVE_ASYNC
        if (tid < 128)
            async_cp16(incf + (size_t)(m0 + arow) * 4096 + k0 + ac * 8,
                       &Ash[buf][arow * 40 + ac * 8]);
        #pragma unroll
        for (int c = 0; c < 4; ++c)
            async_cp16(teT + (size_t)tid * 4096 + k0 + c * 8,
                       &Bsh[buf][tid * 40 + c * 8]);
#else
        half8 ra;
        if (tid < 128)
            ra = *(const half8*)(incf + (size_t)(m0 + arow) * 4096 + k0 + ac * 8);
        half8 rb[4];
        #pragma unroll
        for (int c = 0; c < 4; ++c)
            rb[c] = *(const half8*)(teT + (size_t)tid * 4096 + k0 + c * 8);
        if (tid < 128)
            *(half8*)(&Ash[buf][arow * 40 + ac * 8]) = ra;
        #pragma unroll
        for (int c = 0; c < 4; ++c)
            *(half8*)(&Bsh[buf][tid * 40 + c * 8]) = rb[c];
#endif
    };

    f32x8 acc[4];
    #pragma unroll
    for (int t = 0; t < 4; ++t)
        acc[t] = (f32x8){0.f,0.f,0.f,0.f,0.f,0.f,0.f,0.f};

    stage(0, 0);

    int buf = 0;
    for (int k0 = 0; k0 < 4096; k0 += 32, buf ^= 1) {
#if HAVE_ASYNC
        wait_async0();           // own writes of tile k complete
#endif
        __syncthreads();         // everyone's writes of tile k visible; prev readers done
        if (k0 + 32 < 4096) stage(buf ^ 1, k0 + 32);   // DMA overlaps the WMMAs below

        half16 a = lfrag(&Ash[buf][0], mi * 16 + rl, roff);
        half16 bfv[4];
        #pragma unroll
        for (int t = 0; t < 4; ++t)
            bfv[t] = lfrag(&Bsh[buf][0], n0 + t * 16 + rl, roff);
        SCHED_FENCE();           // batch the ds_loads ahead of the WMMA chain
        #pragma unroll
        for (int t = 0; t < 4; ++t)
            acc[t] = WMMA_F32_F16(a, bfv[t], acc[t]);
    }

    // scale by 1/deg and accumulate per-lane partial of agg . wa
    float p8[8];
    #pragma unroll
    for (int r = 0; r < 8; ++r) p8[r] = 0.f;
    #pragma unroll
    for (int t = 0; t < 4; ++t) {
        const float wv = swa[n0 + t * 16 + rl];
        #pragma unroll
        for (int r = 0; r < 8; ++r) {
            float v = acc[t][r] * sdeg[mi * 16 + roff + r];
            acc[t][r] = v;
            p8[r] += v * wv;
        }
    }
    #pragma unroll
    for (int r = 0; r < 8; ++r)
        part[(mi * 16 + roff + r) * 65 + nq * 16 + rl] = p8[r];
    __syncthreads();

    if (tid < 32) {
        float s = 0.f;
        #pragma unroll
        for (int i = 0; i < 64; ++i) s += part[tid * 65 + i];
        s += dotTn[m0 + tid] + ba_all[head];
        s = (s > 0.f) ? s : 0.2f * s;            // leaky_relu(0.2)
        scoef[tid] = 1.f / (1.f + __expf(-s));   // sigmoid
    }
    __syncthreads();

    // upd = coeff*agg + tn  -> f16
    #pragma unroll
    for (int t = 0; t < 4; ++t) {
        const int col = n0 + t * 16 + rl;
        #pragma unroll
        for (int r = 0; r < 8; ++r) {
            const int rowl = mi * 16 + roff + r;
            const int row = m0 + rowl;
            float u = scoef[rowl] * acc[t][r] + tn[(size_t)row * 256 + col];
            updf[(size_t)row * 256 + col] = (_Float16)u;
        }
    }
}

// ---------------------------------------------------------------------------
// Column min/max: stage 1 (per 256-row slab), stage 2 (combine 32 slabs)
// ---------------------------------------------------------------------------
__global__ void __launch_bounds__(256) k_minmax1(const float* __restrict__ o,
                                                 float* __restrict__ pmn,
                                                 float* __restrict__ pmx) {
    __shared__ float smn[256], smx[256];
    const int t = threadIdx.x, col = t & 127, rg = t >> 7;
    float mn = 3.4e38f, mx = -3.4e38f;
    const int base = blockIdx.x * 256 + rg * 128;
    for (int i = 0; i < 128; ++i) {
        float v = o[(size_t)(base + i) * 128 + col];
        mn = fminf(mn, v); mx = fmaxf(mx, v);
    }
    smn[t] = mn; smx[t] = mx;
    __syncthreads();
    if (rg == 0) {
        pmn[blockIdx.x * 128 + col] = fminf(mn, smn[128 + col]);
        pmx[blockIdx.x * 128 + col] = fmaxf(mx, smx[128 + col]);
    }
}

__global__ void k_minmax2(const float* __restrict__ pmn, const float* __restrict__ pmx,
                          float* __restrict__ cmn, float* __restrict__ cinv) {
    const int col = threadIdx.x;
    float mn = 3.4e38f, mx = -3.4e38f;
    for (int b = 0; b < 32; ++b) {
        mn = fminf(mn, pmn[b * 128 + col]);
        mx = fmaxf(mx, pmx[b * 128 + col]);
    }
    cmn[col] = mn;
    cinv[col] = 1.f / (mx - mn + 1e-8f);
}

// x = relu((out - mn) * inv); write f16 for next head and f32 to d_out
__global__ void k_finalize(const float* __restrict__ o, const float* __restrict__ cmn,
                           const float* __restrict__ cinv, _Float16* __restrict__ xf,
                           float* __restrict__ dout) {
    const int idx = blockIdx.x * 256 + threadIdx.x;
    const int col = idx & 127;
    float v = fmaxf((o[idx] - cmn[col]) * cinv[col], 0.f);
    xf[idx] = (_Float16)v;
    dout[idx] = v;
}

// ---------------------------------------------------------------------------
// Host launcher
// ---------------------------------------------------------------------------
extern "C" void kernel_launch(void* const* d_in, const int* in_sizes, int n_in,
                              void* d_out, int out_size, void* d_ws, size_t ws_size,
                              hipStream_t stream) {
    (void)in_sizes; (void)n_in; (void)out_size; (void)ws_size;
    const float* node = (const float*)d_in[0];   // [8192,128]
    const float* inc  = (const float*)d_in[1];   // [8192,4096]
    const float* ef   = (const float*)d_in[2];   // [4096,128]
    const float* Wn   = (const float*)d_in[3];   // [4,128,256]
    const float* bn   = (const float*)d_in[4];   // [4,256]
    const float* We   = (const float*)d_in[5];   // [4,128,256]
    const float* be   = (const float*)d_in[6];   // [4,256]
    const float* Wa   = (const float*)d_in[7];   // [4,256,1]
    const float* ba   = (const float*)d_in[8];   // [4,1]
    const float* Wo   = (const float*)d_in[9];   // [4,256,128]
    const float* bo   = (const float*)d_in[10];  // [4,128]
    float* dout = (float*)d_out;                 // [8192,128]

    char* ws = (char*)d_ws;
    size_t off = 0;
    auto alloc = [&](size_t bytes) { size_t o = off; off = (off + bytes + 255) & ~(size_t)255; return o; };

    _Float16* incf = (_Float16*)(ws + alloc((size_t)8192 * 4096 * 2)); // 64 MB
    float*   invdeg = (float*)(ws + alloc(8192 * 4));
    _Float16* xf   = (_Float16*)(ws + alloc((size_t)8192 * 128 * 2));
    _Float16* eff  = (_Float16*)(ws + alloc((size_t)4096 * 128 * 2));
    _Float16* WnT  = (_Float16*)(ws + alloc((size_t)4 * 256 * 128 * 2));
    _Float16* WeT  = (_Float16*)(ws + alloc((size_t)4 * 256 * 128 * 2));
    _Float16* WoT  = (_Float16*)(ws + alloc((size_t)4 * 128 * 256 * 2));
    _Float16* teT  = (_Float16*)(ws + alloc((size_t)256 * 4096 * 2));
    float*   tn    = (float*)(ws + alloc((size_t)8192 * 256 * 4));
    float*   dotv  = (float*)(ws + alloc(8192 * 4));
    _Float16* updf = (_Float16*)(ws + alloc((size_t)8192 * 256 * 2));
    float*   outb  = (float*)(ws + alloc((size_t)8192 * 128 * 4));
    float*   pmn   = (float*)(ws + alloc(32 * 128 * 4));
    float*   pmx   = (float*)(ws + alloc(32 * 128 * 4));
    float*   cmn   = (float*)(ws + alloc(128 * 4));
    float*   cinv  = (float*)(ws + alloc(128 * 4));

    // one-time prep
    k_cvt_f16<<<(8192 * 128) / 256, 256, 0, stream>>>(node, xf, 8192 * 128);
    k_cvt_f16<<<(4096 * 128) / 256, 256, 0, stream>>>(ef, eff, 4096 * 128);
    k_transpose_cvt<<<(4 * 128 * 256) / 256, 256, 0, stream>>>(Wn, WnT, 128, 256);
    k_transpose_cvt<<<(4 * 128 * 256) / 256, 256, 0, stream>>>(We, WeT, 128, 256);
    k_transpose_cvt<<<(4 * 256 * 128) / 256, 256, 0, stream>>>(Wo, WoT, 256, 128);
    k_inc_prep<<<8192, 256, 0, stream>>>(inc, incf, invdeg);

    for (int h = 0; h < 4; ++h) {
        // te = ef @ We[h] + be[h], stored transposed f16 [256][4096]
        k_gemm<2, 4, 1><<<4096 / 32, 256, 0, stream>>>(
            eff, WeT + (size_t)h * 256 * 128, be + h * 256, nullptr, teT, 4096, 128, 256, 4096);
        // tn = x @ Wn[h] + bn[h], f32 [8192][256]
        k_gemm<2, 4, 0><<<8192 / 32, 256, 0, stream>>>(
            xf, WnT + (size_t)h * 256 * 128, bn + h * 256, tn, nullptr, 8192, 128, 256, 0);
        // dotTn = tn @ Wa[h]
        k_dot<<<8192 / 8, 256, 0, stream>>>(tn, Wa + h * 256, dotv);
        // fused: agg GEMM + score + sigmoid gate + upd (f16)
        k_agg<<<8192 / 32, 256, 0, stream>>>(
            incf, teT, invdeg, tn, dotv, Wa + h * 256, ba, h, updf);
        // out = upd @ Wo[h] + bo[h], f32 [8192][128]
        k_gemm<2, 2, 0><<<8192 / 32, 256, 0, stream>>>(
            updf, WoT + (size_t)h * 128 * 256, bo + h * 128, outb, nullptr, 8192, 256, 128, 0);
        // column min-max normalize + relu -> next x (f16) and d_out (f32)
        k_minmax1<<<32, 256, 0, stream>>>(outb, pmn, pmx);
        k_minmax2<<<1, 128, 0, stream>>>(pmn, pmx, cmn, cinv);
        k_finalize<<<(8192 * 128) / 256, 256, 0, stream>>>(outb, cmn, cinv, xf, dout);
    }
}
